// SelfAttention_87436944212391
// MI455X (gfx1250) — compile-verified, matrix operand order checked by reading, not run on hardware
//
#include <hip/hip_runtime.h>

typedef __attribute__((ext_vector_type(16))) __bf16 v16bf;
typedef __attribute__((ext_vector_type(8)))  __bf16 v8bf;
typedef __attribute__((ext_vector_type(8)))  float  v8f;

#define B_    8
#define L_    4096
#define DM    65
#define D_    64
#define WAVES 8

union Frag { v16bf v; v8bf h[2]; };

// Load a 16x32 bf16 WMMA fragment from memory where the non-K dimension is
// the row and K is contiguous (A-style: lanes 0-15 = rows, VGPR slots = K;
// lanes 16-31 hold the K+8 / K+24 halves — exact CDNA5 16-bit A layout).
// The same pattern serves B fragments when the source is stored N-major.
__device__ __forceinline__ v16bf load_frag_g(const __bf16* __restrict__ base,
                                             int row0, int stride, int k0, int lane) {
    int m = lane & 15, hh = lane >> 4;
    const __bf16* p = base + (size_t)(row0 + m) * stride + k0 + hh * 8;
    Frag f;
    f.h[0] = *(const v8bf*)(p);        // K +0..7   (or +8..15 in upper half)
    f.h[1] = *(const v8bf*)(p + 16);   // K +16..23 (or +24..31)
    return f.v;
}

__device__ __forceinline__ v8f wmma_bf16(v16bf a, v16bf b, v8f c) {
    return __builtin_amdgcn_wmma_f32_16x16x32_bf16(
        /*neg_a=*/false, a, /*neg_b=*/false, b,
        /*c_mod=*/(short)0, c, /*reuse_a=*/false, /*reuse_b=*/false);
}

// ---------------------------------------------------------------------------
// Kernel 1: fused QKV projection  x[row,65] @ W[65,64] + b  -> bf16 staging.
//   Qb : [B*L, 64] bf16, pre-scaled by log2(e)/sqrt(65) (query-major == the
//        N-major layout a Q^T B-fragment wants)
//   Kb : [B*L, 64] bf16 (row-major == A-fragment layout for K tiles)
//   Vt : [B, 64, L] bf16 (d-major == A-fragment layout for V^T tiles)
// ---------------------------------------------------------------------------
__global__ __launch_bounds__(64)
void qkv_proj_kernel(const float* __restrict__ x,
                     const float* __restrict__ Wq, const float* __restrict__ bq,
                     const float* __restrict__ Wk, const float* __restrict__ bk,
                     const float* __restrict__ Wv, const float* __restrict__ bv,
                     __bf16* __restrict__ Qb, __bf16* __restrict__ Kb,
                     __bf16* __restrict__ Vt) {
    __shared__ float xs[16 * DM];
    const int t    = threadIdx.x;       // output column 0..63
    const int row0 = blockIdx.x * 16;   // 16 token rows per block

    for (int i = t; i < 16 * DM; i += 64) xs[i] = x[(size_t)row0 * DM + i];
    __syncthreads();

    float aq[16], ak[16], av[16];
#pragma unroll
    for (int r = 0; r < 16; ++r) { aq[r] = 0.f; ak[r] = 0.f; av[r] = 0.f; }

    for (int d = 0; d < DM; ++d) {
        float wq = Wq[d * D_ + t], wk = Wk[d * D_ + t], wv = Wv[d * D_ + t];
#pragma unroll
        for (int r = 0; r < 16; ++r) {
            float xv = xs[r * DM + d];
            aq[r] = fmaf(xv, wq, aq[r]);
            ak[r] = fmaf(xv, wk, ak[r]);
            av[r] = fmaf(xv, wv, av[r]);
        }
    }

    const float qscale = 1.4426950408889634f / 8.0622577482985491f; // log2e/sqrt(65)
    const float bqv = bq[t], bkv = bk[t], bvv = bv[t];
    const int b  = row0 / L_;
    const int l0 = row0 % L_;
#pragma unroll
    for (int r = 0; r < 16; ++r) {
        size_t grow = (size_t)row0 + r;
        Qb[grow * D_ + t] = (__bf16)((aq[r] + bqv) * qscale);
        Kb[grow * D_ + t] = (__bf16)(ak[r] + bkv);
        Vt[(size_t)b * D_ * L_ + (size_t)t * L_ + (l0 + r)] = (__bf16)(av[r] + bvv);
    }
}

// ---------------------------------------------------------------------------
// Kernel 2: causal flash attention, transposed-score formulation.
//   S^T = K . Q^T  (A = K tile, B = Q^T)  -> C layout: lane = query, slot = key
//   softmax over keys = in-lane reduction + one xor-16 combine; m/l/scale are
//   per-lane scalars.
//   P (bf16 of exp(S^T)) is already in A/B fragment layout -> no transpose.
//   O^T = V^T . P^T (A = Vt tile, B = P)  -> per-lane 1/l, LDS only in the
//   coalescing epilogue.
// One wave32 per 16-query tile; scalar (SGPR) tile coordinates; no block
// barriers (causal trip counts differ per wave).
// ---------------------------------------------------------------------------
__global__ __launch_bounds__(32 * WAVES)
void flash_attn_kernel(const __bf16* __restrict__ Qb,
                       const __bf16* __restrict__ Kb,
                       const __bf16* __restrict__ Vt,
                       float* __restrict__ out) {
    __shared__ __align__(16) float lds_o[WAVES * 16 * D_];   // epilogue staging

    const int lane  = threadIdx.x & 31;
    const int wave  = __builtin_amdgcn_readfirstlane((int)(threadIdx.x >> 5));
    const int gtile = blockIdx.x * WAVES + wave;
    const int b     = gtile / (L_ / 16);
    const int q0    = (gtile % (L_ / 16)) * 16;

    const __bf16* Qbase = Qb + (size_t)b * L_ * D_;
    const __bf16* Kbase = Kb + (size_t)b * L_ * D_;
    const __bf16* Vbase = Vt + (size_t)b * D_ * L_;

    const int cl = lane & 15;     // this lane's query column
    const int hh = lane >> 4;     // key half within a 16-key C tile

    // Q^T as two resident B fragments (query-major memory == N-major).
    const v16bf qb0 = load_frag_g(Qbase, q0, D_, 0,  lane);
    const v16bf qb1 = load_frag_g(Qbase, q0, D_, 32, lane);

    v8f o0 = {}, o1 = {}, o2 = {}, o3 = {};   // O^T accumulators (lane=query)
    float m = -1e30f, l = 0.f;                // per-lane softmax state
    const int qrow = q0 + cl;

    for (int j0 = 0; j0 <= q0 + 15; j0 += 32) {
        if (j0 + 32 <= q0 + 15)   // scalar-uniform prefetch of next K tile
            __builtin_prefetch(Kbase + (size_t)(j0 + 32 + cl) * D_, 0, 3);

        // K tile as A fragments (row-major == A layout).
        v16bf ka00 = load_frag_g(Kbase, j0,      D_, 0,  lane);
        v16bf ka01 = load_frag_g(Kbase, j0,      D_, 32, lane);
        v16bf ka10 = load_frag_g(Kbase, j0 + 16, D_, 0,  lane);
        v16bf ka11 = load_frag_g(Kbase, j0 + 16, D_, 32, lane);

        // S^T tiles: st0 = keys j0..j0+15, st1 = keys j0+16..j0+31.
        v8f st0 = {}, st1 = {};
        st0 = wmma_bf16(ka00, qb0, st0);
        st0 = wmma_bf16(ka01, qb1, st0);
        st1 = wmma_bf16(ka10, qb0, st1);
        st1 = wmma_bf16(ka11, qb1, st1);

        if (j0 + 31 > q0) {       // ragged causal boundary tile (scalar branch)
#pragma unroll
            for (int r = 0; r < 8; ++r) {
                st0[r] = (j0 + r + 8 * hh      > qrow) ? -1e30f : st0[r];
                st1[r] = (j0 + 16 + r + 8 * hh > qrow) ? -1e30f : st1[r];
            }
        }

        // Per-lane (= per-query) online softmax over this tile's 32 keys.
        float tm = fmaxf(st0[0], st1[0]);
#pragma unroll
        for (int r = 1; r < 8; ++r) tm = fmaxf(tm, fmaxf(st0[r], st1[r]));
        tm = fmaxf(tm, __shfl_xor(tm, 16, 32));   // combine key halves
        float mnew = fmaxf(m, tm);
        float sc   = __builtin_amdgcn_exp2f(m - mnew);
        m = mnew;

        float rs = 0.f;
        Frag pf;
#pragma unroll
        for (int r = 0; r < 8; ++r) {
            float p0 = __builtin_amdgcn_exp2f(st0[r] - mnew);
            float p1 = __builtin_amdgcn_exp2f(st1[r] - mnew);
            rs += p0 + p1;
            pf.v[r]     = (__bf16)p0;   // slots 0..7  : keys kk = r + 8*hh
            pf.v[r + 8] = (__bf16)p1;   // slots 8..15 : keys kk = 16 + r + 8*hh
        }
        rs += __shfl_xor(rs, 16, 32);
        l = l * sc + rs;

#pragma unroll
        for (int r = 0; r < 8; ++r) {
            o0[r] *= sc; o1[r] *= sc; o2[r] *= sc; o3[r] *= sc;
        }

        // O^T += V^T . P^T : A = Vt tiles (d-major), B = P (same registers as
        // the P A-fragment — the S^T C-layout already matches it).
        v16bf va0 = load_frag_g(Vbase, 0,  L_, j0, lane);
        v16bf va1 = load_frag_g(Vbase, 16, L_, j0, lane);
        v16bf va2 = load_frag_g(Vbase, 32, L_, j0, lane);
        v16bf va3 = load_frag_g(Vbase, 48, L_, j0, lane);
        o0 = wmma_bf16(va0, pf.v, o0);
        o1 = wmma_bf16(va1, pf.v, o1);
        o2 = wmma_bf16(va2, pf.v, o2);
        o3 = wmma_bf16(va3, pf.v, o3);
    }

    // Epilogue: scale by 1/l (per-lane), transpose O^T -> O through this
    // wave's private LDS slab, then fully-coalesced b128 stores.
    const float inv = __builtin_amdgcn_rcpf(l);
    float* po = lds_o + wave * (16 * D_) + cl * D_ + 8 * hh;  // [query][d]
    {
        const v8f* oc[4] = { &o0, &o1, &o2, &o3 };
#pragma unroll
        for (int c = 0; c < 4; ++c) {
            const v8f& o = *oc[c];
            float4 lo = { o[0] * inv, o[1] * inv, o[2] * inv, o[3] * inv };
            float4 hi = { o[4] * inv, o[5] * inv, o[6] * inv, o[7] * inv };
            *(float4*)(po + 16 * c)     = lo;   // d = 16c + 8*hh + 0..3
            *(float4*)(po + 16 * c + 4) = hi;   // d = 16c + 8*hh + 4..7
        }
    }
    asm volatile("s_wait_dscnt 0" ::: "memory");  // wave-private slab

    const float* ps = lds_o + wave * (16 * D_);
    float* outb = out + (size_t)b * L_ * D_ + (size_t)q0 * D_;
#pragma unroll
    for (int i = 0; i < 8; ++i) {
        float4 v = *(const float4*)(ps + i * 128 + lane * 4);
        *(float4*)(outb + i * 128 + lane * 4) = v;
    }
}

// ---------------------------------------------------------------------------
extern "C" void kernel_launch(void* const* d_in, const int* in_sizes, int n_in,
                              void* d_out, int out_size, void* d_ws, size_t ws_size,
                              hipStream_t stream) {
    const float* x  = (const float*)d_in[0];
    const float* Wq = (const float*)d_in[1];
    const float* bq = (const float*)d_in[2];
    const float* Wk = (const float*)d_in[3];
    const float* bk = (const float*)d_in[4];
    const float* Wv = (const float*)d_in[5];
    const float* bv = (const float*)d_in[6];
    float* out = (float*)d_out;

    // Workspace: Qb | Kb | Vt, each B*L*64 bf16 (4 MB) -> 12 MB total.
    __bf16* Qb = (__bf16*)d_ws;
    __bf16* Kb = Qb + (size_t)B_ * L_ * D_;
    __bf16* Vt = Kb + (size_t)B_ * L_ * D_;

    qkv_proj_kernel<<<dim3(B_ * L_ / 16), dim3(64), 0, stream>>>(
        x, Wq, bq, Wk, bk, Wv, bv, Qb, Kb, Vt);

    flash_attn_kernel<<<dim3(B_ * (L_ / 16) / WAVES), dim3(32 * WAVES), 0, stream>>>(
        Qb, Kb, Vt, out);
}
